// sparse_hop_9869834846766
// MI455X (gfx1250) — 16-bit and 8-bit matrix operand order repaired and checked by reading, compiled, not run on hardware
//
#include <hip/hip_runtime.h>
#include <hip/hip_bf16.h>

#define HOPS 3
#define BB 4
#define NN 4096
#define CC 256

typedef __attribute__((ext_vector_type(16))) __bf16 v16bf;
typedef __attribute__((ext_vector_type(16))) int    v16i;
typedef __attribute__((ext_vector_type(8)))  float  v8f;

union FragBF { v16bf v; unsigned short u[16]; uint4 q[2]; };
union FragI  { v16i v; uint2 d[8]; uint4 q[4]; };
union Half8  { uint4 q; unsigned short u[8]; };
union Pack16 { uint4 q; unsigned char c[16]; };

#define ONE8 ((unsigned char)0x38)

static __device__ __forceinline__ unsigned short f2bf(float f) {
  unsigned int u = __builtin_bit_cast(unsigned int, f);
  u += 0x7FFFu + ((u >> 16) & 1u);
  return (unsigned short)(u >> 16);
}
static __device__ __forceinline__ float bf2f(unsigned short h) {
  unsigned int u = ((unsigned int)h) << 16;
  return __builtin_bit_cast(float, u);
}
static __device__ __forceinline__ v8f wmma_bf16(v16bf a, v16bf b, v8f c) {
  return __builtin_amdgcn_wmma_f32_16x16x32_bf16(false, a, false, b, (short)0, c,
                                                 false, false);
}
static __device__ __forceinline__ v8f wmma_fp8(v16i a, v16i b, v8f c) {
  return __builtin_amdgcn_wmma_f32_16x16x128_fp8_fp8(a, b, (short)0, c, false, false);
}

static __device__ __forceinline__ v16bf ld_afrag(const unsigned short* s, int row, int ko) {
  FragBF f;
  f.q[0] = *(const uint4*)(s + row * 40 + ko);
  f.q[1] = *(const uint4*)(s + row * 40 + ko + 16);
  return f.v;
}
static __device__ __forceinline__ v16bf ld_bfrag(const unsigned short* s, int n, int half) {
  FragBF f;
  const uint4* p = (const uint4*)(s + n * 40 + half * 8);
  f.q[0] = p[0];
  f.q[1] = p[2];
  return f.v;
}

static __device__ __forceinline__ v16i ld_afrag8(const unsigned char* s, int row, int half) {
  FragI f;
  const unsigned char* p = s + row * 144 + half * 8;
#pragma unroll
  for (int j = 0; j < 8; ++j) f.d[j] = *(const uint2*)(p + 16 * j);
  return f.v;
}
static __device__ __forceinline__ v16i ld_bfrag8(const unsigned char* s, int n, int half) {
  FragI f;
  const unsigned char* p = s + n * 144 + half * 8;
#pragma unroll
  for (int j = 0; j < 8; ++j) f.d[j] = *(const uint2*)(p + 16 * j);
  return f.v;
}

__global__ __launch_bounds__(256) void k_gt(const float* __restrict__ nodes,
                                            const float* __restrict__ cw,
                                            const float* __restrict__ cb,
                                            unsigned short* __restrict__ gt) {
  int idx = blockIdx.x * 256 + threadIdx.x;
  int n   = idx & (NN - 1);
  int o   = (idx >> 12) & (CC - 1);
  int b   = (idx >> 20) & 3;
  int hop = idx >> 22;
  const float* np = nodes + (size_t)b * CC * NN + n;
  const float* wp = cw + (size_t)(hop * CC + o) * CC;
  float acc = cb[hop * CC + o];
#pragma unroll 4
  for (int c = 0; c < CC; ++c) acc += wp[c] * np[(size_t)c * NN];
  gt[idx] = f2bf(acc);
}

__global__ __launch_bounds__(256) void k_gemm_pow1(const float* __restrict__ iw,
                                                   const float* __restrict__ delta,
                                                   unsigned short* __restrict__ hi_p,
                                                   unsigned short* __restrict__ lo_p) {
  __shared__ unsigned char sA[128 * 144];
  __shared__ unsigned char sB[128 * 144];
  const int tid = threadIdx.x;
  const int b   = blockIdx.z;
  const int m0  = blockIdx.y * 128;
  const int n0  = blockIdx.x * 128;
  const float d   = delta[0];
  const float thr = __logf(d) - __logf(1.0f - d);
  const float* iwb = iw + (size_t)b * NN * NN;

  const int lane = tid & 31, wid = tid >> 5;
  const int half = lane >> 4, l16 = lane & 15;
  const int wm0 = (wid & 1) * 64, wn0 = (wid >> 1) * 32;

  v8f acc[4][2] = {};

  for (int k0 = 0; k0 < NN; k0 += 128) {
    if (k0 + 128 < NN) {
      __builtin_prefetch(iwb + (size_t)(m0 + (tid >> 1)) * NN + k0 + 128 + (tid & 1) * 64, 0, 1);
      __builtin_prefetch(iwb + (size_t)(k0 + 128 + (tid >> 3)) * NN + n0 + ((tid & 7) << 4), 0, 1);
    }
    {
      int row = tid >> 1, cc = (tid & 1) * 64;
      const float4* s4 = (const float4*)(iwb + (size_t)(m0 + row) * NN + k0 + cc);
      unsigned char* dst = sA + row * 144 + cc;
#pragma unroll
      for (int j = 0; j < 4; ++j) {
        float4 a = s4[4 * j + 0], bq = s4[4 * j + 1], cq = s4[4 * j + 2], eq = s4[4 * j + 3];
        float t[16] = {a.x,a.y,a.z,a.w, bq.x,bq.y,bq.z,bq.w,
                       cq.x,cq.y,cq.z,cq.w, eq.x,eq.y,eq.z,eq.w};
        Pack16 pk;
#pragma unroll
        for (int i = 0; i < 16; ++i) pk.c[i] = (t[i] >= thr) ? ONE8 : (unsigned char)0;
        *(uint4*)(dst + 16 * j) = pk.q;
      }
    }
#pragma unroll
    for (int it = 0; it < 64; ++it) {
      int idx = it * 256 + tid;
      int kk = idx >> 7, n = idx & 127;
      float f = iwb[(size_t)(k0 + kk) * NN + n0 + n];
      sB[n * 144 + kk] = (f >= thr) ? ONE8 : (unsigned char)0;
    }
    __syncthreads();
    v16i bf[2];
#pragma unroll
    for (int ni = 0; ni < 2; ++ni) bf[ni] = ld_bfrag8(sB, wn0 + ni * 16 + l16, half);
#pragma unroll
    for (int mi = 0; mi < 4; ++mi) {
      v16i af = ld_afrag8(sA, wm0 + mi * 16 + l16, half);
#pragma unroll
      for (int ni = 0; ni < 2; ++ni) acc[mi][ni] = wmma_fp8(af, bf[ni], acc[mi][ni]);
    }
    __syncthreads();
  }
#pragma unroll
  for (int mi = 0; mi < 4; ++mi)
#pragma unroll
    for (int ni = 0; ni < 2; ++ni)
#pragma unroll
      for (int r = 0; r < 8; ++r) {
        int row = m0 + wm0 + mi * 16 + half * 8 + r;
        int col = n0 + wn0 + ni * 16 + l16;
        float v  = acc[mi][ni][r];
        float hv = floorf(v * 0.0625f);
        float lv = v - 16.0f * hv;
        size_t o = (size_t)b * NN * NN + (size_t)row * NN + col;
        hi_p[o] = f2bf(hv);
        lo_p[o] = f2bf(lv);
      }
}

__global__ __launch_bounds__(256) void k_gemm_pow2(const float* __restrict__ iw,
                                                   const float* __restrict__ delta,
                                                   const unsigned short* __restrict__ hi_p,
                                                   const unsigned short* __restrict__ lo_p,
                                                   float* __restrict__ bh2) {
  __shared__ unsigned short sH[64 * 40];
  __shared__ unsigned short sL[64 * 40];
  __shared__ unsigned short sB[128 * 40];
  const int tid = threadIdx.x;
  const int b   = blockIdx.z;
  const int m0  = blockIdx.y * 64;
  const int n0  = blockIdx.x * 128;
  const float d   = delta[0];
  const float thr = __logf(d) - __logf(1.0f - d);
  const float* iwb = iw + (size_t)b * NN * NN;
  const size_t pbase = (size_t)b * NN * NN;

  const int lane = tid & 31, wid = tid >> 5;
  const int half = lane >> 4, l16 = lane & 15;
  const int wm0 = (wid & 1) * 32, wn0 = (wid >> 1) * 32;
  const unsigned short one = 0x3F80u, zero = 0u;

  v8f ah[2][2] = {};
  v8f al[2][2] = {};

  for (int k0 = 0; k0 < NN; k0 += 32) {
    if (k0 + 32 < NN) {
      __builtin_prefetch(hi_p + pbase + (size_t)(m0 + ((tid >> 1) & 63)) * NN + k0 + 32 + (tid & 1) * 16, 0, 1);
      __builtin_prefetch(iwb + (size_t)(k0 + 32 + (tid >> 3)) * NN + n0 + ((tid & 7) << 4), 0, 1);
    }
    {
      int plane = tid >> 7;
      int row = (tid >> 1) & 63;
      int cc  = (tid & 1) * 16;
      const unsigned short* src =
          (plane ? lo_p : hi_p) + pbase + (size_t)(m0 + row) * NN + k0 + cc;
      unsigned short* dst = (plane ? sL : sH) + row * 40 + cc;
      ((uint4*)dst)[0] = ((const uint4*)src)[0];
      *(uint4*)(dst + 8) = *(const uint4*)(src + 8);
    }
#pragma unroll
    for (int it = 0; it < 16; ++it) {
      int idx = it * 256 + tid;
      int kk = idx >> 7, n = idx & 127;
      float f = iwb[(size_t)(k0 + kk) * NN + n0 + n];
      sB[n * 40 + kk] = (f >= thr) ? one : zero;
    }
    __syncthreads();
    v16bf bf[2];
#pragma unroll
    for (int ni = 0; ni < 2; ++ni) bf[ni] = ld_bfrag(sB, wn0 + ni * 16 + l16, half);
#pragma unroll
    for (int mi = 0; mi < 2; ++mi) {
      v16bf hfr = ld_afrag(sH, wm0 + mi * 16 + l16, half * 8);
      v16bf lfr = ld_afrag(sL, wm0 + mi * 16 + l16, half * 8);
#pragma unroll
      for (int ni = 0; ni < 2; ++ni) {
        ah[mi][ni] = wmma_bf16(hfr, bf[ni], ah[mi][ni]);
        al[mi][ni] = wmma_bf16(lfr, bf[ni], al[mi][ni]);
      }
    }
    __syncthreads();
  }
#pragma unroll
  for (int mi = 0; mi < 2; ++mi)
#pragma unroll
    for (int ni = 0; ni < 2; ++ni)
#pragma unroll
      for (int r = 0; r < 8; ++r) {
        int row = m0 + wm0 + mi * 16 + half * 8 + r;
        int col = n0 + wn0 + ni * 16 + l16;
        bh2[pbase + (size_t)row * NN + col] = 16.0f * ah[mi][ni][r] + al[mi][ni][r];
      }
}

__global__ __launch_bounds__(256) void k_stats(const float* __restrict__ iw,
                                               const float* __restrict__ delta,
                                               const unsigned short* __restrict__ hi_p,
                                               const unsigned short* __restrict__ lo_p,
                                               const float* __restrict__ bh2,
                                               float* __restrict__ rowm,
                                               float* __restrict__ rowinvs) {
  int w    = blockIdx.x * 8 + (threadIdx.x >> 5);
  int lane = threadIdx.x & 31;
  int row  = w & (NN - 1);
  int b    = (w >> 12) & 3;
  int hop  = w >> 14;
  const float d   = delta[0];
  const float thr = __logf(d) - __logf(1.0f - d);
  const size_t base = (size_t)(b * NN + row) * NN;

  float m = -3.0e38f, s = 0.0f;
  for (int k = lane; k < NN; k += 32) {
    float x = iw[base + k];
    float bv;
    if (hop == 0)      bv = (x >= thr) ? 1.0f : 0.0f;
    else if (hop == 1) bv = 16.0f * bf2f(hi_p[base + k]) + bf2f(lo_p[base + k]);
    else               bv = bh2[base + k];
    float l  = bv * x;
    float mn = fmaxf(m, l);
    s = s * __expf(m - mn) + __expf(l - mn);
    m = mn;
  }
#pragma unroll
  for (int off = 16; off > 0; off >>= 1) {
    float m2 = __shfl_xor(m, off, 32);
    float s2 = __shfl_xor(s, off, 32);
    float mn = fmaxf(m, m2);
    s = s * __expf(m - mn) + s2 * __expf(m2 - mn);
    m = mn;
  }
  if (lane == 0) { rowm[w] = m; rowinvs[w] = 1.0f / s; }
}

__global__ __launch_bounds__(256) void k_attn(const float* __restrict__ iw,
                                              const float* __restrict__ delta,
                                              const unsigned short* __restrict__ hi_p,
                                              const unsigned short* __restrict__ lo_p,
                                              const float* __restrict__ bh2,
                                              const unsigned short* __restrict__ gt,
                                              const float* __restrict__ rowm,
                                              const float* __restrict__ rowinvs,
                                              float* __restrict__ out) {
  int w    = blockIdx.x * 8 + (threadIdx.x >> 5);
  int lane = threadIdx.x & 31;
  int half = lane >> 4, l16 = lane & 15;
  int ob  = (w & 3) * 64;
  int r0  = ((w >> 2) & 255) * 16;
  int b   = (w >> 10) & 3;
  int hop = w >> 12;
  const float d   = delta[0];
  const float thr = __logf(d) - __logf(1.0f - d);

  int row = r0 + l16;
  const size_t rbase = (size_t)(b * NN + row) * NN;
  const float*          iwr = iw   + rbase;
  const unsigned short* hir = hi_p + rbase;
  const unsigned short* lor = lo_p + rbase;
  const float*          b2r = bh2  + rbase;
  const float mr = rowm   [((hop * 4 + b) << 12) + row];
  const float is = rowinvs[((hop * 4 + b) << 12) + row];
  const unsigned short* gtb = gt + (size_t)(hop * 4 + b) * CC * NN;

  v8f acc[4] = {};
  for (int k0 = 0; k0 < NN; k0 += 32) {
    int ka = k0 + half * 8;
    if (k0 + 32 < NN) {
      __builtin_prefetch(iwr + ka + 32, 0, 1);
      if (hop == 1) {
        __builtin_prefetch(hir + ka + 32, 0, 1);
        __builtin_prefetch(lor + ka + 32, 0, 1);
      } else if (hop == 2) {
        __builtin_prefetch(b2r + ka + 32, 0, 1);
      }
    }
    float x[16];
    {
      const float4* q0 = (const float4*)(iwr + ka);
      const float4* q1 = (const float4*)(iwr + ka + 16);
      float4 a0 = q0[0], a1 = q0[1], a2 = q1[0], a3 = q1[1];
      x[0]=a0.x; x[1]=a0.y; x[2]=a0.z; x[3]=a0.w;
      x[4]=a1.x; x[5]=a1.y; x[6]=a1.z; x[7]=a1.w;
      x[8]=a2.x; x[9]=a2.y; x[10]=a2.z; x[11]=a2.w;
      x[12]=a3.x; x[13]=a3.y; x[14]=a3.z; x[15]=a3.w;
    }
    float bv[16];
    if (hop == 0) {
#pragma unroll
      for (int e = 0; e < 16; ++e) bv[e] = (x[e] >= thr) ? 1.0f : 0.0f;
    } else if (hop == 1) {
      Half8 h0, h1, l0, l1;
      h0.q = *(const uint4*)(hir + ka); h1.q = *(const uint4*)(hir + ka + 16);
      l0.q = *(const uint4*)(lor + ka); l1.q = *(const uint4*)(lor + ka + 16);
#pragma unroll
      for (int e = 0; e < 8; ++e) {
        bv[e]     = 16.0f * bf2f(h0.u[e]) + bf2f(l0.u[e]);
        bv[e + 8] = 16.0f * bf2f(h1.u[e]) + bf2f(l1.u[e]);
      }
    } else {
      const float4* q0 = (const float4*)(b2r + ka);
      const float4* q1 = (const float4*)(b2r + ka + 16);
      float4 a0 = q0[0], a1 = q0[1], a2 = q1[0], a3 = q1[1];
      bv[0]=a0.x; bv[1]=a0.y; bv[2]=a0.z; bv[3]=a0.w;
      bv[4]=a1.x; bv[5]=a1.y; bv[6]=a1.z; bv[7]=a1.w;
      bv[8]=a2.x; bv[9]=a2.y; bv[10]=a2.z; bv[11]=a2.w;
      bv[12]=a3.x; bv[13]=a3.y; bv[14]=a3.z; bv[15]=a3.w;
    }
    FragBF af;
#pragma unroll
    for (int e = 0; e < 16; ++e)
      af.u[e] = f2bf(__expf(bv[e] * x[e] - mr) * is);
#pragma unroll
    for (int ni = 0; ni < 4; ++ni) {
      int o = ob + ni * 16 + l16;
      const uint4* gq = (const uint4*)(gtb + (size_t)o * NN + k0 + half * 8);
      FragBF bf;
      bf.q[0] = gq[0];
      bf.q[1] = gq[2];
      acc[ni] = wmma_bf16(af.v, bf.v, acc[ni]);
    }
  }
#pragma unroll
  for (int ni = 0; ni < 4; ++ni) {
    int o = ob + ni * 16 + l16;
    float* op = out + (size_t)((hop * 4 + b) * CC + o) * NN;
#pragma unroll
    for (int r = 0; r < 8; ++r) op[r0 + half * 8 + r] = acc[ni][r];
  }
}

extern "C" void kernel_launch(void* const* d_in, const int* in_sizes, int n_in,
                              void* d_out, int out_size, void* d_ws, size_t ws_size,
                              hipStream_t stream) {
  const float* iw    = (const float*)d_in[0];
  const float* nodes = (const float*)d_in[1];
  const float* delta = (const float*)d_in[2];
  const float* cw    = (const float*)d_in[3];
  const float* cb    = (const float*)d_in[4];
  float* out = (float*)d_out;

  char* ws = (char*)d_ws;
  const size_t szNN = (size_t)BB * NN * NN;
  unsigned short* hi_p = (unsigned short*)ws;
  unsigned short* lo_p = (unsigned short*)(ws + 2 * szNN);
  float*          bh2  = (float*)(ws + 4 * szNN);
  unsigned short* gt   = (unsigned short*)(ws + 8 * szNN);
  const size_t gtB = (size_t)HOPS * BB * CC * NN * 2;
  float* rowm    = (float*)(ws + 8 * szNN + gtB);
  float* rowinvs = rowm + (size_t)HOPS * BB * NN;

  k_gt<<<(HOPS * BB * CC * NN) / 256, 256, 0, stream>>>(nodes, cw, cb, gt);
  k_gemm_pow1<<<dim3(NN / 128, NN / 128, BB), 256, 0, stream>>>(iw, delta, hi_p, lo_p);
  k_gemm_pow2<<<dim3(NN / 128, NN / 64, BB), 256, 0, stream>>>(iw, delta, hi_p, lo_p, bh2);
  k_stats<<<(HOPS * BB * NN) / 8, 256, 0, stream>>>(iw, delta, hi_p, lo_p, bh2, rowm, rowinvs);
  k_attn<<<(HOPS * BB * (NN / 16) * (CC / 64)) / 8, 256, 0, stream>>>(
      iw, delta, hi_p, lo_p, bh2, gt, rowm, rowinvs, out);
}
